// SO3Convolution_68839735820490
// MI455X (gfx1250) — matrix operand order repaired and checked by reading, not compiled, not run on hardware
//
#include <hip/hip_runtime.h>
#include <hip/hip_bf16.h>

typedef __attribute__((ext_vector_type(16))) __bf16 bf16x16;
typedef __attribute__((ext_vector_type(2)))  __bf16 bf16x2;
typedef __attribute__((ext_vector_type(8)))  float  f32x8;
typedef unsigned short u16;
typedef unsigned int   u32;

#define IRREP 286
#define ROWSTRIDE 18304   // 64 * 286 floats per batch row of x / out

union Frag16 { bf16x16 v; uint4 q[2]; };

__device__ __forceinline__ u16 f2bf(float x) {
    u32 u = __float_as_uint(x);
    u32 r = (u + 0x7FFFu + ((u >> 16) & 1u)) >> 16;   // round-to-nearest-even
    return (u16)r;
}

// Packed f32x2 -> bf16x2 conversion: single v_cvt_pk_bf16_f32.
__device__ __forceinline__ u32 pk2bf(float a, float b) {
#if defined(__has_builtin) && __has_builtin(__builtin_amdgcn_cvt_pk_bf16_f32)
    bf16x2 p = __builtin_amdgcn_cvt_pk_bf16_f32(a, b);
    return __builtin_bit_cast(u32, p);
#else
    u32 r;
    asm("v_cvt_pk_bf16_f32 %0, %1, %2" : "=v"(r) : "v"(a), "v"(b));
    return r;
#endif
}

#if defined(__has_builtin) && __has_builtin(__builtin_amdgcn_global_load_async_to_lds_b128)
#define HAVE_ASYNC_LDS 1
typedef int vi4 __attribute__((vector_size(16)));
typedef __attribute__((address_space(1))) vi4 g_vi4;   // global (AS1)
typedef __attribute__((address_space(3))) vi4 l_vi4;   // LDS (AS3)
#else
#define HAVE_ASYNC_LDS 0
#endif

__device__ __forceinline__ void wait_async0() {
#if defined(__has_builtin) && __has_builtin(__builtin_amdgcn_s_wait_asynccnt)
    __builtin_amdgcn_s_wait_asynccnt(0);
#else
    asm volatile("s_wait_asynccnt 0" ::: "memory");
#endif
}

// ---------------------------------------------------------------------------
// Stage 1: psi[f,g,i] = (1/sqrt(512)) * sum_t D[t,i] w[f,g,t], with the per-l
// 1/sqrt(64 d) scale folded in, repacked per l as bf16 Wl_T[n*K + k],
// k = u*64 + f (K=64d), n = g*d + v (N=64d).  Segment l starts at element
// 4096*OFF[l] in the workspace (OFF is the cumulative sum of d^2).
// ---------------------------------------------------------------------------
__global__ __launch_bounds__(256) void repack_psi_kernel(
        const float* __restrict__ Dm, const float* __restrict__ w,
        u16* __restrict__ ws)
{
    constexpr int DS[6]   = {1, 3, 5, 7, 9, 11};
    constexpr int OFFS[6] = {0, 1, 10, 35, 84, 165};
    const int idx = blockIdx.x * 256 + threadIdx.x;
    if (idx >= 4096 * IRREP) return;

    int l = 0;
#pragma unroll
    for (int t = 1; t < 6; ++t) l = (idx >= 4096 * OFFS[t]) ? t : l;
    const int d   = DS[l];
    const int off = OFFS[l];
    const int K   = 64 * d;
    const int rem = idx - 4096 * off;
    const int n = rem / K;
    const int k = rem - n * K;
    const int u = k >> 6, f = k & 63;
    const int g = n / d,  v = n - g * d;

    const float* __restrict__ wr = w + ((f << 6) + g) * 512;
    const float* __restrict__ dr = Dm + off + u * d + v;
    float acc = 0.f;
#pragma unroll 8
    for (int t = 0; t < 512; ++t) acc += dr[t * IRREP] * wr[t];

    const float scale = rsqrtf(512.0f * 64.0f * (float)d);
    ws[4096 * off + (size_t)n * K + k] = f2bf(acc * scale);
}

// ---------------------------------------------------------------------------
// Stage 2: for degree l (template D, OFF): d GEMMs (one per m) of
// out_m[b, n] = sum_k x_m[b, k] * Wl[k, n], M=4096, K=N=64*D,
// sharing the B matrix across all m.  Block tile 32(M) x 64(N), 256 threads
// = 8 wave32 waves in a 2x4 grid; each wave owns one 16x16 WMMA tile per m.
// ---------------------------------------------------------------------------
template<int D, int OFF>
__global__ __launch_bounds__(256) void so3_block_gemm(
        const float* __restrict__ x, const u16* __restrict__ Wt,
        float* __restrict__ out)
{
    constexpr int K  = 64 * D;
    constexpr int MB = 32;
    constexpr int NB = 64;

    __shared__ __align__(16) u16 As[D * MB * 32];  // [m][b_local][k_local]
    __shared__ __align__(16) u16 Bs[NB * 32];      // [n_local][k_local]

    const int tid  = threadIdx.x;
    const int lane = tid & 31;
    const int wid  = tid >> 5;
    const int wr   = wid >> 2;       // wave row  (0..1) -> 16 M rows each
    const int wc   = wid & 3;        // wave col  (0..3) -> 16 N cols each
    const int b0   = blockIdx.x * MB;
    const int n0   = blockIdx.y * NB;

    f32x8 acc[D];
#pragma unroll
    for (int m = 0; m < D; ++m) acc[m] = (f32x8){0.f,0.f,0.f,0.f,0.f,0.f,0.f,0.f};

    const int half = lane >> 4;
    const int rowL = lane & 15;

    constexpr int steps = K / 32;
    for (int s = 0; s < steps; ++s) {
        const int k0 = s * 32;
        const int u  = k0 >> 6;     // k = u*64 + f
        const int f0 = k0 & 63;     // 0 or 32

        __syncthreads();
        // ---- B tile: Bs[n][k] from Wl_T[n*K + k] ----
        {
            const int rowB = tid >> 2;   // 0..63
            const int ch   = tid & 3;    // 4 chunks of 8 bf16
            const u16* src = Wt + (size_t)(n0 + rowB) * K + k0 + ch * 8;
#if HAVE_ASYNC_LDS
            // CDNA5 async DMA: global -> LDS, tracked by ASYNCcnt, no VGPR stop.
            __builtin_amdgcn_global_load_async_to_lds_b128(
                (g_vi4*)src, (l_vi4*)&Bs[rowB * 32 + ch * 8], 0, 0);
#else
            *(uint4*)&Bs[rowB * 32 + ch * 8] = *(const uint4*)src;
#endif
        }
        // ---- A tile: each thread owns an f-pair so the two bf16 produced per
        //      m are adjacent in LDS -> one packed cvt + one ds_store_b32. ----
#pragma unroll
        for (int i = 0; i < 2; ++i) {
            const int p  = tid + i * 256;     // 0..511 pairs
            const int bl = p >> 4;            // 0..31
            const int fl = (p & 15) * 2;      // even k_local
            const float* s0 = x + (size_t)(b0 + bl) * ROWSTRIDE
                                + (f0 + fl) * IRREP + OFF + u * D;
            const float* s1 = s0 + IRREP;
            if (s + 1 < steps) {
                const int k1 = k0 + 32;
                __builtin_prefetch(x + (size_t)(b0 + bl) * ROWSTRIDE
                                     + ((k1 & 63) + fl) * IRREP + OFF + (k1 >> 6) * D);
            }
#pragma unroll
            for (int m = 0; m < D; ++m)
                *(u32*)&As[m * (MB * 32) + bl * 32 + fl] = pk2bf(s0[m], s1[m]);
        }
#if HAVE_ASYNC_LDS
        wait_async0();
#endif
        __syncthreads();

        // ---- B fragment (shared across all m) ----
        Frag16 bf;
        {
            const u16* bb = &Bs[(wc * 16 + rowL) * 32 + half * 16];
            bf.q[0] = *(const uint4*)(bb);       // K = half*16 + 0..7
            bf.q[1] = *(const uint4*)(bb + 8);   // K = half*16 + 8..15
        }
        // ---- per-m A fragment + WMMA ----
#pragma unroll
        for (int m = 0; m < D; ++m) {
            Frag16 af;
            const u16* ab = &As[m * (MB * 32) + (wr * 16 + rowL) * 32];
            af.q[0] = *(const uint4*)(ab + half * 16);       // K = half*16 + 0..7
            af.q[1] = *(const uint4*)(ab + half * 16 + 8);   // K = half*16 + 8..15
            acc[m] = __builtin_amdgcn_wmma_f32_16x16x32_bf16(
                         false, af.v, false, bf.v, (short)0, acc[m], false, false);
        }
    }

    // ---- epilogue: C/D layout -> out[b, g, OFF + v*D + m] ----
    const int colL  = lane & 15;
    const int mbase = (lane >> 4) * 8;
    const int gcol  = n0 + wc * 16 + colL;
    const int g = gcol / D, v = gcol - g * D;
#pragma unroll
    for (int m = 0; m < D; ++m) {
        float* op = out + (size_t)(b0 + wr * 16 + mbase) * ROWSTRIDE
                        + g * IRREP + OFF + v * D + m;
#pragma unroll
        for (int i = 0; i < 8; ++i)
            op[(size_t)i * ROWSTRIDE] = acc[m][i];
    }
}

extern "C" void kernel_launch(void* const* d_in, const int* in_sizes, int n_in,
                              void* d_out, int out_size, void* d_ws, size_t ws_size,
                              hipStream_t stream) {
    const float* x  = (const float*)d_in[0];   // (4096, 64, 286)
    const float* Dm = (const float*)d_in[1];   // (512, 286)
    const float* w  = (const float*)d_in[2];   // (64, 64, 512)
    float* out = (float*)d_out;                // (4096, 64, 286)
    u16* ws = (u16*)d_ws;                      // 4096*286 bf16 = 2.3 MB

    // Stage 1: fused psi GEMM + scale + per-l transposed bf16 repack
    repack_psi_kernel<<<(4096 * IRREP + 255) / 256, 256, 0, stream>>>(Dm, w, ws);

    // Stage 2: one templated GEMM launch per degree l
    so3_block_gemm<1,   0><<<dim3(128,  1), 256, 0, stream>>>(x, ws + 4096 *   0, out);
    so3_block_gemm<3,   1><<<dim3(128,  3), 256, 0, stream>>>(x, ws + 4096 *   1, out);
    so3_block_gemm<5,  10><<<dim3(128,  5), 256, 0, stream>>>(x, ws + 4096 *  10, out);
    so3_block_gemm<7,  35><<<dim3(128,  7), 256, 0, stream>>>(x, ws + 4096 *  35, out);
    so3_block_gemm<9,  84><<<dim3(128,  9), 256, 0, stream>>>(x, ws + 4096 *  84, out);
    so3_block_gemm<11,165><<<dim3(128, 11), 256, 0, stream>>>(x, ws + 4096 * 165, out);
}
